// ModularRNN_27857157882239
// MI455X (gfx1250) — matrix operand order repaired and checked by reading, not compiled, not run on hardware
//
#include <hip/hip_runtime.h>
#include <math.h>

#define T_LEN 2048
#define M_MOD 64
#define H_DIM 256
#define NWG   64

typedef __attribute__((ext_vector_type(16))) __bf16 v16bf;
typedef __attribute__((ext_vector_type(8)))  float  v8f;

// workspace layout in float units
#define OFF_HA   0u
#define OFF_HB   16384u
#define OFF_E    32768u
#define OFF_S    557056u
#define OFF_BAR  1081344u

__device__ __forceinline__ __bf16 f2bf(float f) { return (__bf16)f; }

__device__ __forceinline__ v8f wmma_bf16(v16bf a, v16bf b, v8f c) {
  return __builtin_amdgcn_wmma_f32_16x16x32_bf16(false, a, false, b, (short)0, c,
                                                 false, false);
}

__device__ __forceinline__ float tanh_fast(float x) {
  // tanh(x) = 1 - 2/(exp(2x)+1)
  const float e = __expf(2.0f * x);
  return 1.0f - 2.0f / (e + 1.0f);
}

// A-operand swizzle: position of logical K index inside the 32-wide chunk so
// lane0 reads positions 0-15 contiguously and lane16 positions 16-31.
__device__ __forceinline__ int aswz(int k) {
  return k;
}

// ---------------------------------------------------------------- init
__global__ void init_ws(float* __restrict__ ws) {
  unsigned i = blockIdx.x * blockDim.x + threadIdx.x;
  if (i < 32768u) ws[OFF_HA + i] = 0.0f;             // hA + hB
  if (i < 524304u) ws[OFF_S + i] = 0.0f;             // s accumulator + barrier words
}

// ------------------------------------------------ bf16 WMMA GEMM (staged)
// C[Mrows x 256] = A[Mrows x 256] * B[256 x 256]; one block per 16-row tile,
// 8 waves x 2 N-tiles each; A/B chunks staged coalesced into LDS in fragment
// layout, converted to bf16 once.
__global__ void __launch_bounds__(256)
gemm_bf16_wmma(const float* __restrict__ A, const float* __restrict__ B,
               float* __restrict__ C, int Mrows) {
  __shared__ __align__(64) __bf16 sA[512];        // 16x32 A chunk, frag-swizzled
  __shared__ __align__(64) __bf16 sB[16 * 512];   // 32x256 B chunk, frag-swizzled

  const int tid  = threadIdx.x;
  const int lane = tid & 31;
  const int wave = tid >> 5;
  const int m0   = blockIdx.x * 16;
  if (m0 >= Mrows) return;
  const int nt0 = wave * 2, nt1 = nt0 + 1;
  const int rc  = lane & 15;

  v8f c0 = {0.f,0.f,0.f,0.f,0.f,0.f,0.f,0.f};
  v8f c1 = {0.f,0.f,0.f,0.f,0.f,0.f,0.f,0.f};

  for (int kt = 0; kt < 8; ++kt) {
    // ---- stage A 16x32 chunk (2 elements/thread, coalesced rows of 32)
#pragma unroll
    for (int r = 0; r < 2; ++r) {
      const int idx = tid + r * 256;            // 0..511
      const int row = idx >> 5, ko = idx & 31;
      const int l = row + ((ko & 16) ? 16 : 0);
      const int e = ko & 15;
      sA[l * 16 + e] = f2bf(A[(size_t)(m0 + row) * H_DIM + kt * 32 + ko]);
    }
    // ---- stage B 32x256 chunk (8 float4/thread, fully coalesced)
#pragma unroll
    for (int j = 0; j < 8; ++j) {
      const int idx = (tid + j * 256) * 4;      // 0..8188
      const int ko = idx >> 8;                  // chunk row 0..31
      const int n  = idx & 255;                 // col, multiple of 4
      const float4 v = *(const float4*)&B[(size_t)(kt * 32 + ko) * H_DIM + n];
      const int nt = n >> 4, cb = n & 15;
      const int lb = (ko >= 16) ? 16 : 0;
      const int e  = ko & 15;
      __bf16* dst = &sB[nt * 512 + lb * 16 + e];
      dst[(cb + 0) * 16] = f2bf(v.x);
      dst[(cb + 1) * 16] = f2bf(v.y);
      dst[(cb + 2) * 16] = f2bf(v.z);
      dst[(cb + 3) * 16] = f2bf(v.w);
    }
    __syncthreads();
    const v16bf a  = *(const v16bf*)&sA[lane * 16];
    const v16bf b0 = *(const v16bf*)&sB[nt0 * 512 + lane * 16];
    const v16bf b1 = *(const v16bf*)&sB[nt1 * 512 + lane * 16];
    c0 = wmma_bf16(a, b0, c0);
    c1 = wmma_bf16(a, b1, c1);
    __syncthreads();
  }
#pragma unroll
  for (int v = 0; v < 8; ++v) {
    const int r = (lane < 16) ? v : (v + 8);
    C[(size_t)(m0 + r) * H_DIM + nt0 * 16 + rc] = c0[v];
    C[(size_t)(m0 + r) * H_DIM + nt1 * 16 + rc] = c1[v];
  }
}

// ---------------------------------------------------- persistent scan
// One workgroup per module; Wx[m],Wh[m] staged in LDS as bf16 WMMA-B fragments.
// Grid-wide sense barrier per timestep through the workspace.
__global__ void __launch_bounds__(256)
modular_rnn_scan(const float* __restrict__ Wx, const float* __restrict__ Wh,
                 const float* __restrict__ bias, const int* __restrict__ conn,
                 const int* __restrict__ in_conn, const int* __restrict__ out_conn,
                 float* __restrict__ ws, float* __restrict__ out) {
  extern __shared__ __align__(128) unsigned char smem[];
  __bf16* wxF     = (__bf16*)smem;               // 128 frags * 512 = 65536 bf16
  __bf16* whF     = wxF + 65536;                 // 65536 bf16
  float*  s_hn    = (float*)(whF + 65536);       // 256 f32 (new hidden state)
  float*  s_b     = s_hn + 256;                  // 256 f32 (bias)
  __bf16* s_inpbf = (__bf16*)(s_b + 256);        // 256 bf16, A-swizzled
  __bf16* s_hbf   = s_inpbf + 256;               // 256 bf16, A-swizzled
  int*    s_msk   = (int*)(s_hbf + 256);         // 64

  const int m    = blockIdx.x;
  const int tid  = threadIdx.x;
  const int lane = tid & 31;
  const int wave = tid >> 5;

  float*       hA    = ws + OFF_HA;
  float*       hB    = ws + OFF_HB;
  const float* e_ts  = ws + OFF_E;
  float*       s_acc = ws + OFF_S;
  unsigned*    bar   = (unsigned*)(ws + OFF_BAR);  // [0]=counter [1]=generation

  // ---- stage weights into LDS in WMMA-B fragment layout (bf16)
  const float* WxM = Wx + (size_t)m * 65536;
  const float* WhM = Wh + (size_t)m * 65536;
  for (int idx = tid; idx < 65536; idx += 256) {
    const int frag = idx >> 9, slot = idx & 511;
    const int l = slot >> 4, e = slot & 15;
    const int kt = frag >> 4, nt = frag & 15;
    const int col = nt * 16 + (l & 15);
    const int k   = kt * 32 + ((l < 16) ? e : (e + 16));
    wxF[idx] = f2bf(WxM[(size_t)k * H_DIM + col]);
    whF[idx] = f2bf(WhM[(size_t)k * H_DIM + col]);
  }
  if (tid < M_MOD) s_msk[tid] = conn[tid * M_MOD + m];  // column m: who feeds me
  s_b[tid]  = bias[m * H_DIM + tid];
  s_hn[tid] = 0.0f;
  s_hbf[tid] = f2bf(0.0f);
  const float inm  = (float)in_conn[m];
  const bool  outm = (out_conn[m] != 0);
  const int   spos = aswz(tid);                  // this thread's swizzled slot
  __syncthreads();

  v16bf zv;
#pragma unroll
  for (int e = 0; e < 16; ++e) zv[e] = f2bf(0.0f);
  const bool act = (lane == 0) || (lane == 16);   // only A row 0 is live
  const int  ag  = lane & 16;                     // A-group offset in chunk
  const int  nt0 = wave * 2, nt1 = nt0 + 1;

  for (int t = 0; t < T_LEN; ++t) {
    const float* hcur  = (t & 1) ? hB : hA;
    float*       hnext = (t & 1) ? hA : hB;

    // ---- inp = sum_i conn[i,m]*h[i] + in_m * (x_t @ in_proj)
    float acc = inm * e_ts[(size_t)t * H_DIM + tid];
    if (t + 1 < T_LEN)
      __builtin_prefetch(e_ts + (size_t)(t + 1) * H_DIM + tid, 0, 1);
#pragma unroll 4
    for (int i = 0; i < M_MOD; ++i)
      if (s_msk[i]) acc += hcur[i * H_DIM + tid];
    s_inpbf[spos] = f2bf(acc);
    __syncthreads();

    // ---- h_new = tanh(inp@Wx + h@Wh + b) via WMMA; wave owns 2 N-tiles
    v8f c0 = {0.f,0.f,0.f,0.f,0.f,0.f,0.f,0.f};
    v8f c1 = {0.f,0.f,0.f,0.f,0.f,0.f,0.f,0.f};
#pragma unroll
    for (int kt = 0; kt < 8; ++kt) {
      const v16bf vin = *(const v16bf*)&s_inpbf[kt * 32 + ag];
      const v16bf vh  = *(const v16bf*)&s_hbf[kt * 32 + ag];
      const v16bf ax = act ? vin : zv;
      const v16bf ah = act ? vh : zv;
      const v16bf bx0 = *(const v16bf*)&wxF[((kt * 16 + nt0) << 9) + lane * 16];
      const v16bf bh0 = *(const v16bf*)&whF[((kt * 16 + nt0) << 9) + lane * 16];
      const v16bf bx1 = *(const v16bf*)&wxF[((kt * 16 + nt1) << 9) + lane * 16];
      const v16bf bh1 = *(const v16bf*)&whF[((kt * 16 + nt1) << 9) + lane * 16];
      c0 = wmma_bf16(ax, bx0, c0);
      c0 = wmma_bf16(ah, bh0, c0);
      c1 = wmma_bf16(ax, bx1, c1);
      c1 = wmma_bf16(ah, bh1, c1);
    }
    if (lane < 16) {                        // C row M=0 lives in VGPR0, lanes 0-15
      const int col0 = nt0 * 16 + lane;
      const int col1 = nt1 * 16 + lane;
      const float y0 = tanh_fast(c0[0] + s_b[col0]);
      const float y1 = tanh_fast(c1[0] + s_b[col1]);
      s_hn[col0] = y0;
      s_hn[col1] = y1;
      hnext[m * H_DIM + col0] = y0;
      hnext[m * H_DIM + col1] = y1;
      if (outm) {
        atomicAdd(&s_acc[(size_t)t * H_DIM + col0], y0);
        atomicAdd(&s_acc[(size_t)t * H_DIM + col1], y1);
      }
    }

    // ---- grid-wide barrier (all 64 persistent workgroups)
    __syncthreads();
    if (tid == 0) {
      __threadfence();
      const unsigned g =
          __hip_atomic_load(&bar[1], __ATOMIC_RELAXED, __HIP_MEMORY_SCOPE_AGENT);
      const unsigned arrived = __hip_atomic_fetch_add(
          &bar[0], 1u, __ATOMIC_ACQ_REL, __HIP_MEMORY_SCOPE_AGENT);
      if (arrived == NWG - 1) {
        __hip_atomic_store(&bar[0], 0u, __ATOMIC_RELAXED, __HIP_MEMORY_SCOPE_AGENT);
        __hip_atomic_fetch_add(&bar[1], 1u, __ATOMIC_RELEASE,
                               __HIP_MEMORY_SCOPE_AGENT);
      } else {
        while (__hip_atomic_load(&bar[1], __ATOMIC_ACQUIRE,
                                 __HIP_MEMORY_SCOPE_AGENT) == g)
          __builtin_amdgcn_s_sleep(2);
      }
      __threadfence();
    }
    __syncthreads();
    s_hbf[spos] = f2bf(s_hn[tid]);          // own hidden state stays in LDS
  }

  // ---- h_final (second output, concatenated after outputs)
  out[(size_t)T_LEN * H_DIM + m * H_DIM + tid] = s_hn[tid];
}

// ---------------------------------------------------------------- launch
extern "C" void kernel_launch(void* const* d_in, const int* in_sizes, int n_in,
                              void* d_out, int out_size, void* d_ws, size_t ws_size,
                              hipStream_t stream) {
  const float* x_seq    = (const float*)d_in[0];
  const float* Wx       = (const float*)d_in[1];
  const float* Wh       = (const float*)d_in[2];
  const float* b        = (const float*)d_in[3];
  const float* in_proj  = (const float*)d_in[4];
  const float* out_proj = (const float*)d_in[5];
  const int*   conn     = (const int*)d_in[6];
  const int*   in_conn  = (const int*)d_in[7];
  const int*   out_conn = (const int*)d_in[8];
  float* out = (float*)d_out;
  float* ws  = (float*)d_ws;

  // 262144 (weight frags) + 1024 (h_new) + 1024 (bias) + 512 + 512 (bf16 vecs)
  // + 256 (mask) = 265472 bytes < 320KB WGP LDS
  const int SMEM = 265472;
  hipFuncSetAttribute((const void*)modular_rnn_scan,
                      hipFuncAttributeMaxDynamicSharedMemorySize, SMEM);

  // 1) zero h buffers, s accumulator, barrier words
  init_ws<<<2049, 256, 0, stream>>>(ws);
  // 2) e = x_seq @ in_proj   (off critical path, WMMA GEMM)
  gemm_bf16_wmma<<<128, 256, 0, stream>>>(x_seq, in_proj, ws + OFF_E, T_LEN);
  // 3) sequential scan: persistent, one workgroup per module
  modular_rnn_scan<<<NWG, 256, SMEM, stream>>>(Wx, Wh, b, conn, in_conn, out_conn,
                                               ws, out);
  // 4) outputs = s @ out_proj
  gemm_bf16_wmma<<<128, 256, 0, stream>>>(ws + OFF_S, out_proj, out, T_LEN);
}
